// StackedFNOBranchNet_55113020342461
// MI455X (gfx1250) — compile-verified, matrix operand order checked by reading, not run on hardware
//
#include <hip/hip_runtime.h>
#include <hip/hip_bf16.h>
#include <math.h>

// ---------------------------------------------------------------------------
// StackedFNOBranchNet for MI455X (gfx1250, wave32, WMMA).
// All dense contractions use v_wmma_f32_16x16x32_f16 (f16 in, f32 acc).
// Partial DFTs are expressed as GEMMs (only 16 W-modes / 32 H-modes survive).
// conv3x3 weights staged in LDS via async global->LDS (gfx1250 path).
// Workspace requirement: ~280 MB.
// ---------------------------------------------------------------------------

typedef __attribute__((ext_vector_type(16))) _Float16 v16h;
typedef __attribute__((ext_vector_type(8)))  float    v8f;

union HF16 { v16h v; _Float16 h[16]; };
union F8x  { v8f  v; float    f[8];  };

#define NN   10       // B*T
#define EE   64       // embedding dim
#define HH   128
#define WW   256
#define HWP  32768    // HH*WW
#define H2   256      // upsampled H
#define W2   512      // upsampled W

#if defined(__has_builtin)
#if __has_builtin(__builtin_amdgcn_global_load_async_to_lds_b32)
#define ASYNC_LDS 1
typedef __attribute__((address_space(1))) int* as1_i32p;
typedef __attribute__((address_space(3))) int* as3_i32p;
#endif
#endif

__device__ __forceinline__ int kmap(int e, int hi) {
  // ISA 7.12.2: 16-bit A/B fragment K index for element e, lane-half hi
  return ((e < 8) ? e : (e + 8)) + (hi << 3);
}

__device__ __forceinline__ v8f vzero() {
  F8x u;
#pragma unroll
  for (int i = 0; i < 8; ++i) u.f[i] = 0.f;
  return u.v;
}

__device__ __forceinline__ v8f wmma_f16(v16h a, v16h b, v8f c) {
  // D = A(16x32 f16) * B(32x16 f16) + C(16x16 f32)
  return __builtin_amdgcn_wmma_f32_16x16x32_f16(false, a, false, b,
                                                (short)0, c, false, false);
}

// Unpredicated fragment gather from f32 memory (all 16 rows valid).
// For A: sm = M-row stride, sk = K stride (lane&15 = M row).
// For B: sm = N-col stride, sk = K stride (lane&15 = N col).
__device__ __forceinline__ v16h gather16(const float* __restrict__ p,
                                         long sm, long sk, float sgn) {
  int lane = threadIdx.x & 31;
  int mm = lane & 15, hi = lane >> 4;
  HF16 r;
#pragma unroll
  for (int e = 0; e < 16; ++e) {
    int k = kmap(e, hi);
    r.h[e] = (_Float16)(sgn * p[(long)mm * sm + (long)k * sk]);
  }
  return r.v;
}

// Row-masked gather: rows >= mcount read row 0 (safe) and are zeroed by select,
// keeping the loads unpredicated (no exec-mask churn).
__device__ __forceinline__ v16h gather16m(const float* __restrict__ p,
                                          long sm, long sk, int mcount, float sgn) {
  int lane = threadIdx.x & 31;
  int mm = lane & 15, hi = lane >> 4;
  bool ok = mm < mcount;
  long mo = ok ? (long)mm * sm : 0;
  HF16 r;
#pragma unroll
  for (int e = 0; e < 16; ++e) {
    int k = kmap(e, hi);
    float v = p[mo + (long)k * sk];
    r.h[e] = (_Float16)(ok ? sgn * v : 0.f);
  }
  return r.v;
}

__device__ __forceinline__ float gelu_exact(float x) {
  return 0.5f * x * (1.0f + erff(x * 0.70710678118654752f));
}

// ---------------------------------------------------------------------------
// prep kernels
// ---------------------------------------------------------------------------

__global__ void zero_kernel(float* p, int n) {
  int t = blockIdx.x * 256 + threadIdx.x;
  if (t < n) p[t] = 0.f;
}

// Pack conv3x3 weights [O=64][I=64][3][3] into WMMA B fragments:
// frag[((tap*2+kc)*4+nt)*32+lane][e] = w[o = nt*16+(lane&15)][i = kmap+32kc][tap]
__global__ void pack_w3_kernel(const float* __restrict__ w, _Float16* __restrict__ dst) {
  int t = blockIdx.x * 256 + threadIdx.x;
  if (t >= 9 * 2 * 4 * 32 * 16) return;
  int e    = t & 15;
  int lane = (t >> 4) & 31;
  int nt   = (t >> 9) & 3;
  int kc   = (t >> 11) & 1;
  int tap  = t >> 12;
  int hi = lane >> 4;
  int i = kmap(e, hi) + (kc << 5);
  int o = (nt << 4) + (lane & 15);
  int ty = tap / 3, tx = tap % 3;
  dst[t] = (_Float16)w[((o * 64 + i) * 3 + ty) * 3 + tx];
}

// Pack DFT matrices into B fragments. table:
// 0 Fw   (K=256 w,  N=32: [0:16]=Re,[16:32]=Im of forward rfft-W, /sqrt(HW))
// 1 FyRe (K=128 y,  N=32 ky32)   2 FyIm   3 FyImNeg
// 4 GyRe (K=32 ky32,N=128 y)     5 GyIm   6 GyImNeg
// 7 Bw   (K=32 [re|im], N=256 w: Hermitian-doubled inverse, /sqrt(HW))
__global__ void pack_dft_kernel(_Float16* __restrict__ dst, int table, int K, int N) {
  int NT = N >> 4;
  int total = (K >> 5) * NT * 512;
  int t = blockIdx.x * 256 + threadIdx.x;
  if (t >= total) return;
  int e    = t & 15;
  int lane = (t >> 4) & 31;
  int nt   = (t >> 9) % NT;
  int kc   = (t >> 9) / NT;
  int hi = lane >> 4;
  int k = kmap(e, hi) + (kc << 5);
  int n = (nt << 4) + (lane & 15);
  const double PI2  = 6.283185307179586476925287;
  const double invs = 1.0 / 181.01933598375618;  // 1/sqrt(128*256)
  double val = 0.0;
  if (table == 0) {
    int kx = (n < 16) ? n : (n - 16);
    double th = PI2 * (double)kx * (double)k / 256.0;
    val = ((n < 16) ? cos(th) : -sin(th)) * invs;
  } else if (table <= 3) {
    int ky = (n < 16) ? n : (96 + n);
    double th = PI2 * (double)k * (double)ky / 128.0;
    val = (table == 1) ? cos(th) : ((table == 2) ? -sin(th) : sin(th));
  } else if (table <= 6) {
    int ky = (k < 16) ? k : (96 + k);
    double th = PI2 * (double)ky * (double)n / 128.0;
    val = (table == 4) ? cos(th) : ((table == 5) ? sin(th) : -sin(th));
  } else {
    if (k < 16) {
      double fk = (k == 0) ? 1.0 : 2.0;
      val = fk * cos(PI2 * (double)k * (double)n / 256.0) * invs;
    } else {
      int kk = k - 16;
      double fk = (kk == 0) ? 1.0 : 2.0;
      val = -fk * sin(PI2 * (double)kk * (double)n / 256.0) * invs;
    }
  }
  dst[t] = (_Float16)(float)val;
}

// ---------------------------------------------------------------------------
// pipeline kernels
// ---------------------------------------------------------------------------

// updim 1x1 conv: [10,2,H,W] -> [10,64,H,W]   (K=2: plain VALU)
__global__ void updim_kernel(const float* __restrict__ sv, const float* __restrict__ w,
                             const float* __restrict__ b, float* __restrict__ xout) {
  long t = (long)blockIdx.x * 256 + threadIdx.x;
  if (t >= (long)NN * HWP) return;
  int n = (int)(t >> 15);
  int hw = (int)(t & (HWP - 1));
  float v0 = sv[(long)(n * 2 + 0) * HWP + hw];
  float v1 = sv[(long)(n * 2 + 1) * HWP + hw];
#pragma unroll 8
  for (int e2 = 0; e2 < EE; ++e2)
    xout[((long)n * EE + e2) * HWP + hw] = w[e2 * 2] * v0 + w[e2 * 2 + 1] * v1 + b[e2];
}

// conv3x3 @256x512 over upsampled input (implicit GEMM via WMMA), accumulating
// instance-norm stats over all pixels, storing only even pixels (nearest resize).
// Weight fragments staged in LDS (73.7 KB of the 320 KB WGP LDS).
__global__ void conv3x3_kernel(const float* __restrict__ xin,
                               const _Float16* __restrict__ wfrag,
                               const float* __restrict__ bias,
                               float* __restrict__ xout,
                               float* __restrict__ stats) {
  __shared__ _Float16 lws[9 * 2 * 4 * 512];   // 36864 halves = 73728 B

  // ---- stage weight fragments into LDS (async global->LDS path) ----
  {
    const int* gsrc = (const int*)wfrag;
    int* ldst = (int*)lws;
    for (int i = threadIdx.x; i < 9 * 2 * 4 * 256; i += 128) {
#if defined(ASYNC_LDS)
      __builtin_amdgcn_global_load_async_to_lds_b32(
          (as1_i32p)(int*)(gsrc + i), (as3_i32p)(ldst + i), 0, 0);
#else
      ldst[i] = gsrc[i];
#endif
    }
#if defined(ASYNC_LDS)
    asm volatile("s_wait_asynccnt 0" ::: "memory");
#endif
    __syncthreads();
  }

  int wave = blockIdx.x * (blockDim.x >> 5) + (threadIdx.x >> 5);
  if (wave >= NN * H2 * (W2 / 16)) return;
  int xs = wave & 31;            // 512/16 strips
  int y  = (wave >> 5) & 255;
  int n  = wave >> 13;
  int lane = threadIdx.x & 31;
  int mm = lane & 15, hi = lane >> 4;
  int xpix = xs * 16 + mm;       // high-res x of this A-lane

  __builtin_prefetch(&xin[((long)n * EE) * HWP + (long)(y >> 1) * WW], 0, 3);

  v8f acc[4];
#pragma unroll
  for (int q = 0; q < 4; ++q) acc[q] = vzero();

  const v16h* bfr = (const v16h*)lws;
  for (int ty = 0; ty < 3; ++ty) {
    int ys = y + ty - 1;
    if (ys < 0 || ys >= H2) continue;       // uniform across wave
    int ylo = ys >> 1;
    for (int tx = 0; tx < 3; ++tx) {
      int xsrc = xpix + tx - 1;
      bool xvalid = (xsrc >= 0) && (xsrc < W2);
      int xc  = xsrc < 0 ? 0 : (xsrc > (W2 - 1) ? (W2 - 1) : xsrc);
      int xlo = xc >> 1;
      int tap = ty * 3 + tx;
      const float* src = xin + ((long)n * EE * HH + ylo) * WW + xlo;
#pragma unroll
      for (int kc = 0; kc < 2; ++kc) {
        HF16 a;
#pragma unroll
        for (int e = 0; e < 16; ++e) {
          int k = kmap(e, hi) + (kc << 5);
          float v = src[(long)k * HWP];        // unpredicated load
          a.h[e] = (_Float16)(xvalid ? v : 0.f);
        }
        // preload all 4 B fragments from LDS, then issue the WMMA group
        v16h b0 = bfr[((tap * 2 + kc) * 4 + 0) * 32 + lane];
        v16h b1 = bfr[((tap * 2 + kc) * 4 + 1) * 32 + lane];
        v16h b2 = bfr[((tap * 2 + kc) * 4 + 2) * 32 + lane];
        v16h b3 = bfr[((tap * 2 + kc) * 4 + 3) * 32 + lane];
        acc[0] = wmma_f16(a.v, b0, acc[0]);
        acc[1] = wmma_f16(a.v, b1, acc[1]);
        acc[2] = wmma_f16(a.v, b2, acc[2]);
        acc[3] = wmma_f16(a.v, b3, acc[3]);
      }
    }
  }

  int ylo = y >> 1;
  bool ykeep = (y & 1) == 0;
#pragma unroll
  for (int nt = 0; nt < 4; ++nt) {
    int o = nt * 16 + mm;
    float bv = bias[o];
    F8x c; c.v = acc[nt];
    float s1 = 0.f, s2 = 0.f;
#pragma unroll
    for (int v = 0; v < 8; ++v) {
      float val = c.f[v] + bv;
      s1 += val; s2 += val * val;
      int xp = xs * 16 + v + 8 * hi;        // parity of xp == parity of v
      if (ykeep && ((v & 1) == 0))
        xout[(((long)(n * EE + o)) * HH + ylo) * WW + (xp >> 1)] = val;
    }
    atomicAdd(&stats[(n * EE + o) * 2 + 0], s1);
    atomicAdd(&stats[(n * EE + o) * 2 + 1], s2);
  }
}

// instance-norm (biased var over 256x512) + exact GELU
__global__ void normgelu_kernel(const float* __restrict__ xin,
                                const float* __restrict__ stats,
                                float* __restrict__ xout) {
  long t = (long)blockIdx.x * 256 + threadIdx.x;
  if (t >= (long)NN * EE * HWP) return;
  int nc = (int)(t >> 15);
  const float inv = 1.0f / (float)(H2 * W2);
  float mean = stats[nc * 2 + 0] * inv;
  float var  = stats[nc * 2 + 1] * inv - mean * mean;
  float val = (xin[t] - mean) * rsqrtf(var + 1e-5f);
  xout[t] = gelu_exact(val);
}

// partial rfft over W: rows (n,e,y)=81920, K=256, N=32 (16 Re + 16 Im)
__global__ void rfftw_kernel(const float* __restrict__ x, const _Float16* __restrict__ fw,
                             float* __restrict__ c1re, float* __restrict__ c1im) {
  int wave = blockIdx.x * (blockDim.x >> 5) + (threadIdx.x >> 5);
  if (wave >= 5120) return;
  long r0 = (long)wave * 16;
  int lane = threadIdx.x & 31;
  int mm = lane & 15, hi = lane >> 4;
  v8f aR = vzero(), aI = vzero();
  const v16h* bfr = (const v16h*)fw;
  const float* abase = x + r0 * WW;
#pragma unroll
  for (int kc = 0; kc < 8; ++kc) {
    v16h a  = gather16(abase + kc * 32, WW, 1, 1.f);
    v16h bR = bfr[(kc * 2 + 0) * 32 + lane];
    v16h bI = bfr[(kc * 2 + 1) * 32 + lane];
    aR = wmma_f16(a, bR, aR);
    aI = wmma_f16(a, bI, aI);
  }
  F8x cr, ci; cr.v = aR; ci.v = aI;
#pragma unroll
  for (int v = 0; v < 8; ++v) {
    long row = r0 + v + 8 * hi;
    c1re[row * 16 + mm] = cr.f[v];
    c1im[row * 16 + mm] = ci.f[v];
  }
}

// complex fft over H for the 16 retained kx: per (n,e): [16kx x 128y] -> [16kx x 32ky]
__global__ void ffty_kernel(const float* __restrict__ c1re, const float* __restrict__ c1im,
                            const _Float16* __restrict__ fyre, const _Float16* __restrict__ fyim,
                            const _Float16* __restrict__ fyimn,
                            float* __restrict__ c2re, float* __restrict__ c2im) {
  int wave = blockIdx.x * (blockDim.x >> 5) + (threadIdx.x >> 5);
  if (wave >= NN * EE) return;
  int ne = wave;
  int lane = threadIdx.x & 31;
  int mm = lane & 15, hi = lane >> 4;
  v8f accR[2], accI[2];
#pragma unroll
  for (int q = 0; q < 2; ++q) { accR[q] = vzero(); accI[q] = vzero(); }
  const v16h* bR  = (const v16h*)fyre;
  const v16h* bI  = (const v16h*)fyim;
  const v16h* bIN = (const v16h*)fyimn;
  const float* baseR = c1re + (long)ne * 2048;  // [y*16 + kx]
  const float* baseI = c1im + (long)ne * 2048;
#pragma unroll
  for (int kc = 0; kc < 4; ++kc) {
    v16h ar = gather16(baseR + kc * 32 * 16, 1, 16, 1.f);
    v16h ai = gather16(baseI + kc * 32 * 16, 1, 16, 1.f);
#pragma unroll
    for (int nt = 0; nt < 2; ++nt) {
      int bi = (kc * 2 + nt) * 32 + lane;
      accR[nt] = wmma_f16(ar, bR[bi],  accR[nt]);
      accR[nt] = wmma_f16(ai, bIN[bi], accR[nt]);
      accI[nt] = wmma_f16(ar, bI[bi],  accI[nt]);
      accI[nt] = wmma_f16(ai, bR[bi],  accI[nt]);
    }
  }
#pragma unroll
  for (int nt = 0; nt < 2; ++nt) {
    F8x cr, ci; cr.v = accR[nt]; ci.v = accI[nt];
#pragma unroll
    for (int v = 0; v < 8; ++v) {
      int kx = v + 8 * hi;
      long idx = ((long)ne * 16 + kx) * 32 + nt * 16 + mm;
      c2re[idx] = cr.f[v];
      c2im[idx] = ci.f[v];
    }
  }
}

// per-mode complex channel mixing: for each of 512 modes, [10 x 64c] x [64c x 64c]
__global__ void spectral_kernel(const float* __restrict__ c2re, const float* __restrict__ c2im,
                                const float* __restrict__ wr, const float* __restrict__ wi,
                                int layer,
                                float* __restrict__ s2re, float* __restrict__ s2im) {
  int wave = blockIdx.x * (blockDim.x >> 5) + (threadIdx.x >> 5);
  if (wave >= 512) return;
  int my = wave & 15;            // kx
  int mx = (wave >> 4) & 15;     // ky within corner
  int corner = wave >> 8;
  int jj = corner * 16 + mx;     // ky32 index
  int lane = threadIdx.x & 31;
  int mm = lane & 15, hi = lane >> 4;

  v8f accR[4], accI[4];
#pragma unroll
  for (int q = 0; q < 4; ++q) { accR[q] = vzero(); accI[q] = vzero(); }

  const float* abR = c2re + my * 32 + jj;   // (n,i) strides: 32768, 512
  const float* abI = c2im + my * 32 + jj;
  long wbase = (long)(layer * 2 + corner) * 64 * 64 * 256 + (long)mx * 16 + my;

  for (int kc = 0; kc < 2; ++kc) {
    v16h ar = gather16m(abR + (long)kc * 32 * 512, 32768, 512, 10, 1.f);
    v16h ai = gather16m(abI + (long)kc * 32 * 512, 32768, 512, 10, 1.f);
#pragma unroll
    for (int nt = 0; nt < 4; ++nt) {
      const float* wrp = wr + wbase + (long)kc * 32 * 16384 + (long)nt * 16 * 256;
      const float* wip = wi + wbase + (long)kc * 32 * 16384 + (long)nt * 16 * 256;
      v16h br  = gather16(wrp, 256, 16384, 1.f);
      v16h bi  = gather16(wip, 256, 16384, 1.f);
      v16h biN = gather16(wip, 256, 16384, -1.f);
      accR[nt] = wmma_f16(ar, br,  accR[nt]);
      accR[nt] = wmma_f16(ai, biN, accR[nt]);
      accI[nt] = wmma_f16(ar, bi,  accI[nt]);
      accI[nt] = wmma_f16(ai, br,  accI[nt]);
    }
  }
#pragma unroll
  for (int nt = 0; nt < 4; ++nt) {
    F8x cr, ci; cr.v = accR[nt]; ci.v = accI[nt];
#pragma unroll
    for (int v = 0; v < 8; ++v) {
      int n = v + 8 * hi;
      if (n < 10) {
        int o = nt * 16 + mm;
        long idx = ((long)(n * EE + o) * 16 + my) * 32 + jj;
        s2re[idx] = cr.f[v];
        s2im[idx] = ci.f[v];
      }
    }
  }
}

// inverse fft over H: per (n,o): [16kx x 32ky] -> [16kx x 128y], store interleaved re|im
__global__ void iffty_kernel(const float* __restrict__ s2re, const float* __restrict__ s2im,
                             const _Float16* __restrict__ gyre, const _Float16* __restrict__ gyim,
                             const _Float16* __restrict__ gyimn,
                             float* __restrict__ c3ri) {
  int wave = blockIdx.x * (blockDim.x >> 5) + (threadIdx.x >> 5);
  if (wave >= NN * EE) return;
  int no = wave;
  int lane = threadIdx.x & 31;
  int mm = lane & 15, hi = lane >> 4;
  const v16h* bR  = (const v16h*)gyre;
  const v16h* bI  = (const v16h*)gyim;
  const v16h* bIN = (const v16h*)gyimn;
  v16h ar = gather16(s2re + (long)no * 512, 32, 1, 1.f);
  v16h ai = gather16(s2im + (long)no * 512, 32, 1, 1.f);
#pragma unroll
  for (int nt = 0; nt < 8; ++nt) {
    v8f accR = vzero(), accI = vzero();
    int bi = nt * 32 + lane;
    accR = wmma_f16(ar, bR[bi],  accR);
    accR = wmma_f16(ai, bIN[bi], accR);
    accI = wmma_f16(ar, bI[bi],  accI);
    accI = wmma_f16(ai, bR[bi],  accI);
    F8x cr, ci; cr.v = accR; ci.v = accI;
#pragma unroll
    for (int v = 0; v < 8; ++v) {
      int kx = v + 8 * hi;
      int yv = nt * 16 + mm;
      long base = ((long)no * HH + yv) * 32;
      c3ri[base + kx]      = cr.f[v];
      c3ri[base + 16 + kx] = ci.f[v];
    }
  }
}

// Hermitian inverse rfft over W: rows (n,o,y)=81920, K=32 (re|im), N=256
__global__ void irfftw_kernel(const float* __restrict__ c3ri, const _Float16* __restrict__ bw,
                              float* __restrict__ spec) {
  int wave = blockIdx.x * (blockDim.x >> 5) + (threadIdx.x >> 5);
  if (wave >= 5120) return;
  long r0 = (long)wave * 16;
  int lane = threadIdx.x & 31;
  int mm = lane & 15, hi = lane >> 4;
  const v16h* bfr = (const v16h*)bw;
  v16h a = gather16(c3ri + r0 * 32, 32, 1, 1.f);
#pragma unroll
  for (int nt = 0; nt < 16; ++nt) {
    v8f acc = vzero();
    acc = wmma_f16(a, bfr[nt * 32 + lane], acc);
    F8x c; c.v = acc;
#pragma unroll
    for (int v = 0; v < 8; ++v)
      spec[(r0 + v + 8 * hi) * WW + nt * 16 + mm] = c.f[v];
  }
}

// mix: gelu(conv1x1(x, cw, cb) + spec)  -> xnext
__global__ void mix_kernel(const float* __restrict__ x, const float* __restrict__ spec,
                           const float* __restrict__ cw, const float* __restrict__ cb,
                           int layer, float* __restrict__ xout) {
  int wave = blockIdx.x * (blockDim.x >> 5) + (threadIdx.x >> 5);
  if (wave >= NN * (HWP / 16)) return;
  int n = wave >> 11;
  int hw0 = (wave & 2047) * 16;
  int lane = threadIdx.x & 31;
  int mm = lane & 15, hi = lane >> 4;
  const float* abase = x + (long)n * EE * HWP + hw0;  // pixel stride 1, i stride HWP
  v16h a0 = gather16(abase,             1, HWP, 1.f);
  v16h a1 = gather16(abase + 32L * HWP, 1, HWP, 1.f);
#pragma unroll
  for (int nt = 0; nt < 4; ++nt) {
    v8f acc = vzero();
    const float* cwb = cw + (long)(layer * EE + nt * 16) * EE;  // o stride 64, i stride 1
    v16h b0 = gather16(cwb,      64, 1, 1.f);
    v16h b1 = gather16(cwb + 32, 64, 1, 1.f);
    acc = wmma_f16(a0, b0, acc);
    acc = wmma_f16(a1, b1, acc);
    F8x c; c.v = acc;
    int o = nt * 16 + mm;
    float bv = cb[layer * EE + o];
#pragma unroll
    for (int v = 0; v < 8; ++v) {
      long idx = (long)(n * EE + o) * HWP + hw0 + v + 8 * hi;
      float val = c.f[v] + bv + spec[idx];
      xout[idx] = gelu_exact(val);
    }
  }
}

// downdim 1x1 conv: 64 -> 2 (VALU)
__global__ void downdim_kernel(const float* __restrict__ x, const float* __restrict__ w,
                               const float* __restrict__ b, float* __restrict__ out) {
  long t = (long)blockIdx.x * 256 + threadIdx.x;
  if (t >= (long)NN * HWP) return;
  int n = (int)(t >> 15);
  int hw = (int)(t & (HWP - 1));
#pragma unroll
  for (int cc = 0; cc < 2; ++cc) {
    float acc = b[cc];
#pragma unroll 16
    for (int i = 0; i < EE; ++i)
      acc += x[((long)n * EE + i) * HWP + hw] * w[cc * EE + i];
    out[((long)n * 2 + cc) * HWP + hw] = acc;
  }
}

// ---------------------------------------------------------------------------
// host launcher
// ---------------------------------------------------------------------------

extern "C" void kernel_launch(void* const* d_in, const int* in_sizes, int n_in,
                              void* d_out, int out_size, void* d_ws, size_t ws_size,
                              hipStream_t stream) {
  const float* sensor   = (const float*)d_in[0];
  const float* updim_w  = (const float*)d_in[1];
  const float* updim_b  = (const float*)d_in[2];
  const float* smooth_w = (const float*)d_in[3];
  const float* smooth_b = (const float*)d_in[4];
  const float* fno_wr   = (const float*)d_in[5];
  const float* fno_wi   = (const float*)d_in[6];
  const float* fno_cw   = (const float*)d_in[7];
  const float* fno_cb   = (const float*)d_in[8];
  const float* dd_w     = (const float*)d_in[9];
  const float* dd_b     = (const float*)d_in[10];

  char* wsp = (char*)d_ws;
  size_t off = 0;
  auto alloc = [&](size_t bytes) -> void* {
    void* p = wsp + off;
    off = (off + bytes + 255) & ~(size_t)255;
    return p;
  };

  const size_t SZX = (size_t)NN * EE * HWP;       // 20,971,520 floats
  float* xA    = (float*)alloc(SZX * 4);
  float* xB    = (float*)alloc(SZX * 4);
  float* spec  = (float*)alloc(SZX * 4);
  float* c1re  = (float*)alloc((size_t)NN * EE * HH * 16 * 4);
  float* c1im  = (float*)alloc((size_t)NN * EE * HH * 16 * 4);
  float* c2re  = (float*)alloc((size_t)NN * EE * 16 * 32 * 4);
  float* c2im  = (float*)alloc((size_t)NN * EE * 16 * 32 * 4);
  float* s2re  = (float*)alloc((size_t)NN * EE * 16 * 32 * 4);
  float* s2im  = (float*)alloc((size_t)NN * EE * 16 * 32 * 4);
  float* c3ri  = (float*)alloc((size_t)NN * EE * HH * 32 * 4);
  float* stats = (float*)alloc((size_t)NN * EE * 2 * 4);
  _Float16* w3f   = (_Float16*)alloc(9 * 2 * 4 * 512 * 2);
  _Float16* fw    = (_Float16*)alloc(8 * 2 * 512 * 2);
  _Float16* fyre  = (_Float16*)alloc(4 * 2 * 512 * 2);
  _Float16* fyim  = (_Float16*)alloc(4 * 2 * 512 * 2);
  _Float16* fyimn = (_Float16*)alloc(4 * 2 * 512 * 2);
  _Float16* gyre  = (_Float16*)alloc(1 * 8 * 512 * 2);
  _Float16* gyim  = (_Float16*)alloc(1 * 8 * 512 * 2);
  _Float16* gyimn = (_Float16*)alloc(1 * 8 * 512 * 2);
  _Float16* bw    = (_Float16*)alloc(1 * 16 * 512 * 2);
  (void)ws_size; (void)n_in; (void)in_sizes; (void)out_size;

  // --- prep ---
  zero_kernel<<<5, 256, 0, stream>>>(stats, NN * EE * 2);
  pack_w3_kernel<<<144, 256, 0, stream>>>(smooth_w, w3f);
  pack_dft_kernel<<<32, 256, 0, stream>>>(fw,    0, 256, 32);
  pack_dft_kernel<<<16, 256, 0, stream>>>(fyre,  1, 128, 32);
  pack_dft_kernel<<<16, 256, 0, stream>>>(fyim,  2, 128, 32);
  pack_dft_kernel<<<16, 256, 0, stream>>>(fyimn, 3, 128, 32);
  pack_dft_kernel<<<16, 256, 0, stream>>>(gyre,  4, 32, 128);
  pack_dft_kernel<<<16, 256, 0, stream>>>(gyim,  5, 32, 128);
  pack_dft_kernel<<<16, 256, 0, stream>>>(gyimn, 6, 32, 128);
  pack_dft_kernel<<<32, 256, 0, stream>>>(bw,    7, 32, 256);

  // --- front-end ---
  updim_kernel<<<1280, 256, 0, stream>>>(sensor, updim_w, updim_b, xA);
  conv3x3_kernel<<<20480, 128, 0, stream>>>(xA, w3f, smooth_b, xB, stats);
  normgelu_kernel<<<81920, 256, 0, stream>>>(xB, stats, xA);

  // --- FNO layers ---
  float* cur = xA;
  float* nxt = xB;
  for (int l = 0; l < 4; ++l) {
    rfftw_kernel<<<1280, 128, 0, stream>>>(cur, fw, c1re, c1im);
    ffty_kernel<<<160, 128, 0, stream>>>(c1re, c1im, fyre, fyim, fyimn, c2re, c2im);
    spectral_kernel<<<128, 128, 0, stream>>>(c2re, c2im, fno_wr, fno_wi, l, s2re, s2im);
    iffty_kernel<<<160, 128, 0, stream>>>(s2re, s2im, gyre, gyim, gyimn, c3ri);
    irfftw_kernel<<<1280, 128, 0, stream>>>(c3ri, bw, spec);
    mix_kernel<<<5120, 128, 0, stream>>>(cur, spec, fno_cw, fno_cb, l, nxt);
    float* tmp = cur; cur = nxt; nxt = tmp;
  }

  // --- back-end ---
  downdim_kernel<<<1280, 256, 0, stream>>>(cur, dd_w, dd_b, (float*)d_out);
}
